// PhasePairs_20985210208770
// MI455X (gfx1250) — 8-bit matrix operand order repaired and checked by reading, compiled, not run on hardware
//
#include <hip/hip_runtime.h>
#include <stdint.h>

#define G_ 512
#define T_ 128
#define P_ 64
#define B_ 128

#define APITCH 80

typedef __attribute__((ext_vector_type(8))) int v8i;

__global__ __launch_bounds__(256) void pack_bits_kernel(
    const int* __restrict__ ap, const int* __restrict__ bp,
    const int* __restrict__ pv,
    uint32_t* __restrict__ apack, uint32_t* __restrict__ bpack,
    uint32_t* __restrict__ pvpack)
{
    const int NA = G_ * T_ * P_ / 4;
    const int NP = B_ * P_ / 4;
    int i = blockIdx.x * 256 + threadIdx.x;
    if (i < NA) {
        int4 a = reinterpret_cast<const int4*>(ap)[i];
        apack[i] = (uint32_t)((a.x & 1) | ((a.y & 1) << 8) |
                              ((a.z & 1) << 16) | ((a.w & 1) << 24));
        int4 b = reinterpret_cast<const int4*>(bp)[i];
        bpack[i] = (uint32_t)((b.x & 1) | ((b.y & 1) << 8) |
                              ((b.z & 1) << 16) | ((b.w & 1) << 24));
        if (i < NP) {
            int4 p = reinterpret_cast<const int4*>(pv)[i];
            pvpack[i] = (uint32_t)((p.x & 1) | ((p.y & 1) << 8) |
                                   ((p.z & 1) << 16) | ((p.w & 1) << 24));
        }
    }
}

__device__ inline uint32_t term_pack(int a, int b) {
    const int gg = (a + b) & 7;
    const int sa = (((a + 1) & 7) < 4) ? 1 : -1;
    const int sb = (((b + 1) & 7) < 4) ? 1 : -1;
    const int sg = (((gg + 1) & 7) < 4) ? 1 : -1;
    const int ka = a & 3, kb = b & 3, kg = gg & 3;
    const int y0 = 1 + (ka == 0 ? sa : 0) + (kb == 0 ? sb : 0) - (kg == 0 ? sg : 0);
    const int y1 =     (ka == 1 ? sa : 0) + (kb == 1 ? sb : 0) - (kg == 1 ? sg : 0);
    const int y2 =     (ka == 2 ? sa : 0) + (kb == 2 ? sb : 0) - (kg == 2 ? sg : 0);
    const int y3 =     (ka == 3 ? sa : 0) + (kb == 3 ? sb : 0) - (kg == 3 ? sg : 0);
    return (uint32_t)(y0 & 0xFF) | ((uint32_t)(y1 & 0xFF) << 8) |
           ((uint32_t)(y2 & 0xFF) << 16) | ((uint32_t)(y3 & 0xFF) << 24);
}

__global__ __launch_bounds__(256) void phase_prod_kernel(
    const int* __restrict__ alpha, const int* __restrict__ beta,
    const int* __restrict__ counts,
    const uint8_t* __restrict__ apack, const uint8_t* __restrict__ bpack,
    const uint8_t* __restrict__ pvpack,
    float* __restrict__ out)
{
    __shared__ __align__(16) uint8_t sA[T_ * APITCH];
    __shared__ __align__(16) uint8_t sB[T_ * APITCH];
    __shared__ int      sComb[T_];
    __shared__ uint32_t sTab[64];

    const int g    = blockIdx.x;
    const int tid  = threadIdx.x;
    const int bT   = tid >> 5;
    const int lane = tid & 31;
    const int b0   = bT << 4;
    const int n    = lane & 15;
    const int h    = lane >> 4;

    const int* __restrict__ alpha_g = alpha + g * T_;
    const int* __restrict__ beta_g  = beta  + g * T_;

    {
        const uint8_t* gA = apack + (size_t)g * T_ * P_;
        const uint8_t* gB = bpack + (size_t)g * T_ * P_;
#pragma unroll
        for (int s = 0; s < 2; ++s) {
            const int c = tid + s * 256;
            const uint32_t dst = (uint32_t)((c >> 2) * APITCH + (c & 3) * 16);
            const uint32_t ldsA = (uint32_t)(uint64_t)(uintptr_t)&sA[dst];
            const uint32_t ldsB = (uint32_t)(uint64_t)(uintptr_t)&sB[dst];
            const uint64_t srcA = (uint64_t)(uintptr_t)(gA + (size_t)c * 16);
            const uint64_t srcB = (uint64_t)(uintptr_t)(gB + (size_t)c * 16);
            asm volatile("global_load_async_to_lds_b128 %0, %1, off"
                         :: "v"(ldsA), "v"(srcA) : "memory");
            asm volatile("global_load_async_to_lds_b128 %0, %1, off"
                         :: "v"(ldsB), "v"(srcB) : "memory");
        }
        if (tid < T_) {
            sComb[tid] = (alpha_g[tid] << 3) | beta_g[tid];
        } else if (tid < T_ + 64) {
            const int i = tid - T_;
            sTab[i] = term_pack(i >> 3, i & 7);
        }
        asm volatile("s_wait_asynccnt 0x0" ::: "memory");
        __syncthreads();
    }

    v8i bfrag{};
#pragma unroll
    for (int r = 0; r < 8; ++r) {
        const int K0 = ((r >> 1) << 4) + ((r & 1) << 2) + (h << 3);
        bfrag[r] = *reinterpret_cast<const int*>(pvpack + (size_t)(b0 + n) * P_ + K0);
    }

    const int cnt = counts[g];

    uint32_t p0 = 1u, p1 = 0u, p2 = 0u, p3 = 0u;
    const v8i zero = {};

    for (int tt = 0; tt < 8; ++tt) {
        const int t0 = tt << 4;

        v8i aA{}, aB{};
        const uint8_t* arow = sA + (t0 + n) * APITCH;
        const uint8_t* brow = sB + (t0 + n) * APITCH;
#pragma unroll
        for (int r = 0; r < 8; ++r) {
            const int k0 = ((r >> 1) << 4) + ((r & 1) << 2) + (h << 3);
            aA[r] = *reinterpret_cast<const int*>(arow + k0);
            aB[r] = *reinterpret_cast<const int*>(brow + k0);
        }

        v8i cA = __builtin_amdgcn_wmma_i32_16x16x64_iu8(false, aA, false, bfrag,
                                                        zero, false, false);
        v8i cB = __builtin_amdgcn_wmma_i32_16x16x64_iu8(false, aB, false, bfrag,
                                                        zero, false, false);

#pragma unroll
        for (int r = 0; r < 8; ++r) {
            const int t = t0 + (h << 3) + r;
            if (t < cnt) {
                const int idx = sComb[t] ^ ((cA[r] & 1) << 5) ^ ((cB[r] & 1) << 2);
                const uint32_t tp = sTab[idx];
                const uint32_t y0 = (uint32_t)((int)(tp << 24) >> 24);
                const uint32_t y1 = (uint32_t)((int)(tp << 16) >> 24);
                const uint32_t y2 = (uint32_t)((int)(tp << 8) >> 24);
                const uint32_t y3 = (uint32_t)((int)tp >> 24);
                const uint32_t n0 = p0 * y0 - p1 * y3 - p2 * y2 - p3 * y1;
                const uint32_t n1 = p0 * y1 + p1 * y0 - p2 * y3 - p3 * y2;
                const uint32_t n2 = p0 * y2 + p1 * y1 + p2 * y0 - p3 * y3;
                const uint32_t n3 = p0 * y3 + p1 * y2 + p2 * y1 + p3 * y0;
                p0 = n0; p1 = n1; p2 = n2; p3 = n3;
            }
        }
    }

    const uint32_t q0 = __shfl_down(p0, 16, 32);
    const uint32_t q1 = __shfl_down(p1, 16, 32);
    const uint32_t q2 = __shfl_down(p2, 16, 32);
    const uint32_t q3 = __shfl_down(p3, 16, 32);

    if (h == 0) {
        const uint32_t f0 = p0 * q0 - p1 * q3 - p2 * q2 - p3 * q1;
        const uint32_t f1 = p0 * q1 + p1 * q0 - p2 * q3 - p3 * q2;
        const uint32_t f2 = p0 * q2 + p1 * q1 + p2 * q0 - p3 * q3;
        const uint32_t f3 = p0 * q3 + p1 * q2 + p2 * q1 + p3 * q0;
        float4 o;
        o.x = (float)(int)f0;
        o.y = (float)(int)f1;
        o.z = (float)(int)f2;
        o.w = (float)(int)f3;
        reinterpret_cast<float4*>(out)[(size_t)(b0 + n) * G_ + g] = o;
    }
}

extern "C" void kernel_launch(void* const* d_in, const int* in_sizes, int n_in,
                              void* d_out, int out_size, void* d_ws, size_t ws_size,
                              hipStream_t stream)
{
    (void)in_sizes; (void)n_in; (void)out_size; (void)ws_size;

    const int* alpha        = (const int*)d_in[0];
    const int* alpha_params = (const int*)d_in[1];
    const int* beta         = (const int*)d_in[2];
    const int* beta_params  = (const int*)d_in[3];
    const int* counts       = (const int*)d_in[4];
    const int* param_vals   = (const int*)d_in[5];

    uint8_t* apack  = (uint8_t*)d_ws;
    uint8_t* bpack  = apack + (size_t)G_ * T_ * P_;
    uint8_t* pvpack = bpack + (size_t)G_ * T_ * P_;

    const int NA = G_ * T_ * P_ / 4;
    pack_bits_kernel<<<(NA + 255) / 256, 256, 0, stream>>>(
        alpha_params, beta_params, param_vals,
        (uint32_t*)apack, (uint32_t*)bpack, (uint32_t*)pvpack);

    phase_prod_kernel<<<G_, 256, 0, stream>>>(
        alpha, beta, counts, apack, bpack, pvpack, (float*)d_out);
}
